// CausalSelfAttention_82532091560478
// MI455X (gfx1250) — matrix operand order repaired and checked by reading, not compiled, not run on hardware
//
#include <hip/hip_runtime.h>

typedef __attribute__((ext_vector_type(16))) _Float16 v16h;
typedef __attribute__((ext_vector_type(8)))  _Float16 v8h;
typedef __attribute__((ext_vector_type(8)))  float    v8f;
typedef __attribute__((ext_vector_type(4)))  float    f32x4;
typedef __attribute__((ext_vector_type(4)))  unsigned int u32x4;
typedef __attribute__((ext_vector_type(4)))  int      i32x4;
typedef __attribute__((ext_vector_type(8)))  int      i32x8;

#define BB    2
#define TT    2048
#define CDIM  1024
#define HEADS 16
#define HD    64
#define MTOK  (BB*TT)          // 4096

static __device__ __forceinline__ v16h cat8(v8h lo, v8h hi) {
    return __builtin_shufflevector(lo, hi, 0,1,2,3,4,5,6,7,8,9,10,11,12,13,14,15);
}

static __device__ __forceinline__ v8f wmma_f16(v16h a, v16h b, v8f c) {
    return __builtin_amdgcn_wmma_f32_16x16x32_f16(false, a, false, b, (short)0, c, false, false);
}

// Low 32 bits of a generic pointer to __shared__ == LDS byte offset.
static __device__ __forceinline__ unsigned lds_byte_addr(const void* p) {
    return (unsigned)(unsigned long long)p;
}

// ---------------------------------------------------------------------------
// TDM: issue tensor_load_to_lds for a 2D f16 tile with LDS row padding.
//   rows      : tile_dim1 (number of rows)
//   rowElems  : tile_dim0 (contiguous f16 per row)
//   tensorDim0/rowStride : tensor row length / stride (f16 units)
//   tensorDim1: number of rows in whole tensor
//   padIntLog : pad_interval code (row bytes = 8 * 2^padIntLog)
//   padAmt    : pad_amount code (pad DWORDs - 1)
// ---------------------------------------------------------------------------
static __device__ __forceinline__ void tdm_load_tile_f16(
    const _Float16* gsrc, unsigned ldsAddr,
    unsigned rowElems, unsigned rows,
    unsigned tensorDim0, unsigned tensorDim1, unsigned rowStride,
    unsigned padIntLog, unsigned padAmt)
{
    unsigned long long ga = (unsigned long long)gsrc;
    u32x4 g0;
    g0.x = 1u;                                       // count=1, user mode
    g0.y = ldsAddr;                                  // lds_addr [63:32]
    g0.z = (unsigned)(ga & 0xFFFFFFFFu);             // global_addr [95:64]
    g0.w = (unsigned)(ga >> 32) | (2u << 30);        // global_addr hi + type=2

    i32x8 g1;
    g1.s0 = (int)((1u << 16)                         // data_size = 2 bytes
                | (1u << 20)                         // pad_enable
                | (padIntLog << 22)                  // pad_interval
                | (padAmt << 25));                   // pad_amount
    g1.s1 = (int)((tensorDim0 & 0xFFFFu) << 16);     // tensor_dim0[15:0] @63:48
    g1.s2 = (int)((tensorDim0 >> 16) | ((tensorDim1 & 0xFFFFu) << 16));
    g1.s3 = (int)((tensorDim1 >> 16) | (rowElems << 16));  // tile_dim0 @127:112
    g1.s4 = (int)rows;                               // tile_dim1 (tile_dim2=0)
    g1.s5 = (int)rowStride;                          // tensor_dim0_stride low
    g1.s6 = 0;                                       // stride hi + dim1_stride lo
    g1.s7 = 0;

    i32x4 gz = {0, 0, 0, 0};
#if __clang_major__ >= 23
    __builtin_amdgcn_tensor_load_to_lds(g0, g1, gz, gz, (i32x8){0,0,0,0,0,0,0,0}, 0);
#else
    __builtin_amdgcn_tensor_load_to_lds(g0, g1, gz, gz, 0);
#endif
}

// ---------------------------------------------------------------------------
// GEMM: C[M,N] = A[M,K](f32) * W[K,N](f32) + bias[N]
// MODE 0: store f32 to Cout[M,N]
// MODE 1: scatter into Q[B,H,T,D], K[B,H,T,D], Vt[B,H,D,T] as f16 (qkv split)
// Block: 256 threads (8 waves), tile 128(M) x 64(N), K step 64 (2 sub-steps).
// ---------------------------------------------------------------------------
template <int MODE>
__global__ __launch_bounds__(256)
void gemm_wmma_kernel(const float* __restrict__ A, const float* __restrict__ W,
                      const float* __restrict__ bias, float* __restrict__ Cout,
                      _Float16* __restrict__ Qf, _Float16* __restrict__ Kf,
                      _Float16* __restrict__ Vtf, int M, int N, int K)
{
    constexpr int LDS_S = 72;                 // padded halfword stride (144B)
    __shared__ _Float16 As[128 * LDS_S];      // 18 KiB
    __shared__ _Float16 Bs[64 * LDS_S];       //  9 KiB, transposed [n][k]

    const int tid  = threadIdx.x;
    const int lane = tid & 31;
    const int wave = tid >> 5;
    const int wm   = wave & 3;                // 4 waves along M
    const int wn   = wave >> 2;               // 2 waves along N
    const int ll   = lane & 15;
    const int lh   = lane >> 4;

    const int m0 = blockIdx.y * 128;
    const int n0 = blockIdx.x * 64;

    // cooperative-load assignments (K-slice width 64)
    const int arow = tid >> 1;                // 0..127
    const int acol = (tid & 1) * 32;          // 0 / 32
    const int brow = tid >> 2;                // k 0..63
    const int bcol = (tid & 3) * 16;          // n 0..48

    v8f acc[2][2] = {};

    for (int kk = 0; kk < K; kk += 64) {
        __syncthreads();
        { // A tile: rows m0+arow, k = kk+acol .. +31 (f32 -> f16)
            const float* src = A + (size_t)(m0 + arow) * K + kk + acol;
#pragma unroll
            for (int q = 0; q < 2; q++) {
                f32x4 a0 = *(const f32x4*)(src + q * 16 + 0);
                f32x4 a1 = *(const f32x4*)(src + q * 16 + 4);
                f32x4 a2 = *(const f32x4*)(src + q * 16 + 8);
                f32x4 a3 = *(const f32x4*)(src + q * 16 + 12);
                v8h h0, h1;
#pragma unroll
                for (int i = 0; i < 4; i++) {
                    h0[i] = (_Float16)a0[i];  h0[i+4] = (_Float16)a1[i];
                    h1[i] = (_Float16)a2[i];  h1[i+4] = (_Float16)a3[i];
                }
                *(v8h*)&As[arow * LDS_S + acol + q * 16]     = h0;
                *(v8h*)&As[arow * LDS_S + acol + q * 16 + 8] = h1;
            }
        }
        { // B tile transposed: W[kk+brow][n0+bcol..+15] -> Bs[n][k]
            const float* src = W + (size_t)(kk + brow) * N + n0 + bcol;
#pragma unroll
            for (int q = 0; q < 4; q++) {
                f32x4 b0 = *(const f32x4*)(src + q * 4);
#pragma unroll
                for (int i = 0; i < 4; i++)
                    Bs[(bcol + q * 4 + i) * LDS_S + brow] = (_Float16)b0[i];
            }
        }
        __syncthreads();

#pragma unroll
        for (int ks = 0; ks < 64; ks += 32) {
            v16h af[2], bf[2];
#pragma unroll
            for (int mi = 0; mi < 2; mi++) {  // A frag: K runs at ks+8*lh, ks+16+8*lh
                const _Float16* p = &As[(wm * 32 + mi * 16 + ll) * LDS_S + ks + lh * 8];
                af[mi] = cat8(*(const v8h*)p, *(const v8h*)(p + 16));
            }
#pragma unroll
            for (int ni = 0; ni < 2; ni++) {  // B frag: K runs at ks+8*lh, ks+16+8*lh
                const _Float16* p = &Bs[(wn * 32 + ni * 16 + ll) * LDS_S + ks + lh * 8];
                bf[ni] = cat8(*(const v8h*)p, *(const v8h*)(p + 16));
            }
#pragma unroll
            for (int mi = 0; mi < 2; mi++)
#pragma unroll
                for (int ni = 0; ni < 2; ni++)
                    acc[mi][ni] = wmma_f16(af[mi], bf[ni], acc[mi][ni]);
        }
    }

    // epilogue: D layout = row (r + 8*lh), col (lane&15)
#pragma unroll
    for (int mi = 0; mi < 2; mi++)
#pragma unroll
        for (int ni = 0; ni < 2; ni++) {
            const int col = n0 + wn * 32 + ni * 16 + ll;
            const float bv = bias[col];
#pragma unroll
            for (int r = 0; r < 8; r++) {
                const int row = m0 + wm * 32 + mi * 16 + r + 8 * lh;
                const float val = acc[mi][ni][r] + bv;
                if (MODE == 0) {
                    Cout[(size_t)row * N + col] = val;
                } else {
                    const int b = row >> 11, t = row & 2047;
                    const int which = col >> 10, cc = col & 1023;
                    const int h = cc >> 6, d = cc & 63;
                    const size_t bh = (size_t)(b * HEADS + h);
                    const _Float16 hv = (_Float16)val;
                    if      (which == 0) Qf [(bh * TT + t) * HD + d] = hv;
                    else if (which == 1) Kf [(bh * TT + t) * HD + d] = hv;
                    else                 Vtf[(bh * HD + d) * TT + t] = hv;
                }
            }
        }
}

// ---------------------------------------------------------------------------
// Flash attention: per wave one 16-query tile; key blocks of 32 staged into
// LDS by the Tensor Data Mover (wave 0 issues two tensor_load_to_lds + waits
// tensorcnt), online softmax, P round-trip through per-wave LDS staging.
// Block: 256 threads (8 waves) -> 128 queries per (b,h).
// ---------------------------------------------------------------------------
__global__ __launch_bounds__(256)
void attn_wmma_kernel(const _Float16* __restrict__ Qf, const _Float16* __restrict__ Kf,
                      const _Float16* __restrict__ Vtf, float* __restrict__ Y)
{
    __shared__ _Float16 Ks[32 * 72];          // [key][d], stride 72 (144B)
    __shared__ _Float16 Vs[64 * 40];          // [d][key], stride 40 (80B)
    __shared__ _Float16 Ps[8][16 * 40];       // per-wave P staging [row][key]

    const int tid  = threadIdx.x;
    const int lane = tid & 31;
    const int wave = tid >> 5;
    const int ll   = lane & 15;
    const int lh   = lane >> 4;

    const int bh = blockIdx.y;                // 0..31
    const int b  = bh >> 4, h = bh & 15;
    const int q0 = blockIdx.x * 128 + wave * 16;

    const _Float16* Qb = Qf  + (size_t)bh * TT * HD;
    const _Float16* Kb = Kf  + (size_t)bh * TT * HD;
    const _Float16* Vb = Vtf + (size_t)bh * HD * TT;

    const unsigned ksAddr = lds_byte_addr(&Ks[0]);
    const unsigned vsAddr = lds_byte_addr(&Vs[0]);

    // Q fragments (A-layout), d = 0..31 and 32..63
    v16h qf[2];
    {
        const _Float16* qp = Qb + (size_t)(q0 + ll) * HD + lh * 8;
        qf[0] = cat8(*(const v8h*)qp,        *(const v8h*)(qp + 16));
        qf[1] = cat8(*(const v8h*)(qp + 32), *(const v8h*)(qp + 48));
    }

    float mrow[8], lrow[8];
    v8f o[4] = {};
#pragma unroll
    for (int r = 0; r < 8; r++) { mrow[r] = -1e30f; lrow[r] = 0.0f; }

    const int nkb = blockIdx.x * 4 + 4;       // key blocks needed for this query block

    for (int jb = 0; jb < nkb; jb++) {
        const int j = jb * 32;
        __syncthreads();                      // all waves done with previous tile
        if (wave == 0) {
            // K tile: 32 rows x 64 f16, row = 128B = 32 DW -> pad_interval 4;
            // pad +16B (4 DW, code 3) -> LDS stride 72 halfwords.
            tdm_load_tile_f16(Kb + (size_t)j * HD, ksAddr,
                              /*rowElems=*/HD, /*rows=*/32,
                              /*tensorDim0=*/HD, /*tensorDim1=*/TT, /*rowStride=*/HD,
                              /*padIntLog=*/4, /*padAmt=*/3);
            // V tile: 64 rows x 32 f16, row = 64B = 16 DW -> pad_interval 3;
            // pad +16B -> LDS stride 40 halfwords.
            tdm_load_tile_f16(Vb + j, vsAddr,
                              /*rowElems=*/32, /*rows=*/HD,
                              /*tensorDim0=*/TT, /*tensorDim1=*/HD, /*rowStride=*/TT,
                              /*padIntLog=*/3, /*padAmt=*/3);
            __builtin_amdgcn_s_wait_tensorcnt(0);
        }
        __syncthreads();                      // tiles visible to all waves
        if (j > q0 + 15) continue;            // causally irrelevant for this wave

        // S = Q * K^T : two 16x16 key tiles, d split in two K=32 chunks
        v8f s0 = {}, s1 = {};
        {
            const _Float16* p0 = &Ks[ll * 72 + lh * 8];
            v16h b00 = cat8(*(const v8h*)p0,        *(const v8h*)(p0 + 16));
            v16h b01 = cat8(*(const v8h*)(p0 + 32), *(const v8h*)(p0 + 48));
            const _Float16* p1 = &Ks[(16 + ll) * 72 + lh * 8];
            v16h b10 = cat8(*(const v8h*)p1,        *(const v8h*)(p1 + 16));
            v16h b11 = cat8(*(const v8h*)(p1 + 32), *(const v8h*)(p1 + 48));
            s0 = wmma_f16(qf[0], b00, s0); s0 = wmma_f16(qf[1], b01, s0);
            s1 = wmma_f16(qf[0], b10, s1); s1 = wmma_f16(qf[1], b11, s1);
        }

        const float scale = 0.125f;           // 1/sqrt(64)
        const bool needMask = (j + 31 > q0);
        float fac[8];
#pragma unroll
        for (int r = 0; r < 8; r++) {
            const int qrow = q0 + r + 8 * lh;
            float v0 = s0[r] * scale, v1 = s1[r] * scale;
            if (needMask) {
                if (j + ll      > qrow) v0 = -1e30f;
                if (j + 16 + ll > qrow) v1 = -1e30f;
            }
            float rm = fmaxf(v0, v1);
#pragma unroll
            for (int off = 8; off >= 1; off >>= 1) rm = fmaxf(rm, __shfl_xor(rm, off, 16));
            const float mn = fmaxf(mrow[r], rm);
            const float f  = __expf(mrow[r] - mn);
            const float p0 = __expf(v0 - mn);
            const float p1 = __expf(v1 - mn);
            float rs = p0 + p1;
#pragma unroll
            for (int off = 8; off >= 1; off >>= 1) rs += __shfl_xor(rs, off, 16);
            lrow[r] = lrow[r] * f + rs;
            mrow[r] = mn;
            fac[r]  = f;
            Ps[wave][(r + 8 * lh) * 40 + ll]      = (_Float16)p0;
            Ps[wave][(r + 8 * lh) * 40 + 16 + ll] = (_Float16)p1;
        }
#pragma unroll
        for (int ni = 0; ni < 4; ni++)
#pragma unroll
            for (int r = 0; r < 8; r++) o[ni][r] *= fac[r];

        // P (A-layout, K=32 keys) from same-wave LDS staging
        v16h pf;
        {
            const _Float16* pp = &Ps[wave][ll * 40 + lh * 8];
            pf = cat8(*(const v8h*)pp, *(const v8h*)(pp + 16));
        }
#pragma unroll
        for (int ni = 0; ni < 4; ni++) {      // O += P * V, d in 4 N-tiles
            const _Float16* vp = &Vs[(ni * 16 + ll) * 40 + lh * 8];
            v16h vf = cat8(*(const v8h*)vp, *(const v8h*)(vp + 16));
            o[ni] = wmma_f16(pf, vf, o[ni]);
        }
    }

    // normalize + write y[B,T,C]
#pragma unroll
    for (int r = 0; r < 8; r++) {
        const float inv = 1.0f / lrow[r];
        const int t = q0 + r + 8 * lh;
#pragma unroll
        for (int ni = 0; ni < 4; ni++) {
            const int c = h * HD + ni * 16 + ll;
            Y[((size_t)b * TT + t) * CDIM + c] = o[ni][r] * inv;
        }
    }
}

// ---------------------------------------------------------------------------
extern "C" void kernel_launch(void* const* d_in, const int* in_sizes, int n_in,
                              void* d_out, int out_size, void* d_ws, size_t ws_size,
                              hipStream_t stream) {
    const float* x      = (const float*)d_in[0];
    const float* W_attn = (const float*)d_in[1];
    const float* b_attn = (const float*)d_in[2];
    const float* W_proj = (const float*)d_in[3];
    const float* b_proj = (const float*)d_in[4];
    float* out = (float*)d_out;

    // workspace layout (bytes): Q f16 8 MiB | K f16 8 MiB | Vt f16 8 MiB | y f32 16 MiB
    char* ws = (char*)d_ws;
    _Float16* Qf  = (_Float16*)(ws);
    _Float16* Kf  = (_Float16*)(ws + 8388608);
    _Float16* Vtf = (_Float16*)(ws + 16777216);
    float*    Y   = (float*)   (ws + 25165824);

    // 1) qkv = x @ W_attn + b_attn, scattered to Q/K/Vt (f16)
    dim3 g1(3 * CDIM / 64, MTOK / 128);       // 48 x 32
    gemm_wmma_kernel<1><<<g1, 256, 0, stream>>>(x, W_attn, b_attn, nullptr,
                                                Qf, Kf, Vtf, MTOK, 3 * CDIM, CDIM);

    // 2) flash causal attention -> y[B,T,C] (f32)
    dim3 g2(TT / 128, BB * HEADS);            // 16 x 32
    attn_wmma_kernel<<<g2, 256, 0, stream>>>(Qf, Kf, Vtf, Y);

    // 3) out = y @ W_proj + b_proj
    dim3 g3(CDIM / 64, MTOK / 128);           // 16 x 32
    gemm_wmma_kernel<0><<<g3, 256, 0, stream>>>(Y, W_proj, b_proj, out,
                                                nullptr, nullptr, nullptr, MTOK, CDIM, CDIM);
}
